// DoubleInputPoincareMLP_34548716929073
// MI455X (gfx1250) — matrix operand order repaired and checked by reading, not compiled, not run on hardware
//
#include <hip/hip_runtime.h>

typedef __attribute__((ext_vector_type(16))) _Float16 v16h;
typedef __attribute__((ext_vector_type(8)))  float    v8f;
typedef __attribute__((ext_vector_type(4)))  float    v4f;

#define THREADS 256
#define WAVES_PER_BLOCK 8
#define TILES_PER_WAVE 4

__device__ __forceinline__ float rsum_pair(float x) { return x + __shfl_xor(x, 16, 32); }
__device__ __forceinline__ float atanh_fast(float m) { return 0.5f * __logf((1.0f + m) / (1.0f - m)); }
// tanh for n >= 0, overflow-safe form (exp(2n)->inf gives exactly 1.0)
__device__ __forceinline__ float tanh_pos(float n) { float e = __expf(2.0f * n); return 1.0f - 2.0f / (e + 1.0f); }
__device__ __forceinline__ unsigned pack2h(float a, float b) {
  union { _Float16 h[2]; unsigned u; } t;
  t.h[0] = (_Float16)a; t.h[1] = (_Float16)b; return t.u;
}
union BFrag { v16h h; unsigned u[8]; };

// Feature-major dataflow, per wave = 16 samples:
//   D[out_feat x sample] = A(W^T) x B(logmap(x)^T)   via v_wmma_f32_16x16x32_f16
// C/D layout: lane holds (M = 8*laneHi + v  [out feature in tile], N = laneLo [sample]).
// B layout:   lane holds (N = laneLo [sample], K = 16*laneHi + j [in feature]).
// A layout:   lane holds (M = laneLo, halves j -> K = 16*laneHi + j).
__global__ __launch_bounds__(THREADS)
void poincare_mlp_wmma(const float* __restrict__ x1, const float* __restrict__ x2,
                       const float* __restrict__ Wc1, const float* __restrict__ Wc2,
                       const float* __restrict__ bc,  const float* __restrict__ W0,
                       const float* __restrict__ b0,  const float* __restrict__ W1,
                       const float* __restrict__ b1,  float* __restrict__ out,
                       int nRows)
{
  // 20 A-side fragments of W^T (f16), 1KB each: Wc1[0..3], Wc2[4..7], W0[8..15], W1[16..19]
  __shared__ __align__(64) _Float16 fragLDS[20 * 512];
  __shared__ __align__(16) float    biasLDS[160];   // bc[0..63], b0[64..127], b1[128..159]

  const int tid = threadIdx.x;

  // ---- one-time per block: pack W^T into WMMA A-fragment layout in LDS ----
  for (int slot = tid; slot < 20 * 32; slot += THREADS) {
    const int f = slot >> 5, ln = slot & 31;
    const float* Wp; int ld, ot, kb;
    if (f < 4)       { Wp = Wc1; ld = 64; ot = f;            kb = 0; }
    else if (f < 8)  { Wp = Wc2; ld = 64; ot = f - 4;        kb = 0; }
    else if (f < 16) { Wp = W0;  ld = 64; ot = (f - 8) & 3;  kb = ((f - 8) >> 2) * 32; }
    else             { Wp = W1;  ld = 32; ot = (f - 16) & 1; kb = ((f - 16) >> 1) * 32; }
    const int col = 16 * ot + (ln & 15);   // output feature (A-matrix row M)
    const int kro = kb + 16 * (ln >> 4);
    v16h fr;
    #pragma unroll
    for (int j = 0; j < 16; ++j) {
      const int K = kro + j;
      fr[j] = (_Float16)Wp[K * ld + col];  // W^T[o][k] = W[k][o]
    }
    *(v16h*)&fragLDS[f * 512 + ln * 16] = fr;
  }
  for (int i = tid; i < 64; i += THREADS) { biasLDS[i] = bc[i]; biasLDS[64 + i] = b0[i]; }
  for (int i = tid; i < 32; i += THREADS) biasLDS[128 + i] = b1[i];
  __syncthreads();

  const int lane = tid & 31, laneLo = lane & 15, laneHi = lane >> 4;
  const int wv = tid >> 5;
  const int nTiles = (nRows + 15) >> 4;

  // ---------- helpers ----------
  auto loadA = [&](int f) -> v16h { return *(const v16h*)&fragLDS[f * 512 + lane * 16]; };

  auto sumsq64 = [&](const float (&xh)[4][8]) -> float {
    float s = 0.f;
    #pragma unroll
    for (int ot = 0; ot < 4; ++ot)
      #pragma unroll
      for (int v = 0; v < 8; ++v) s += xh[ot][v] * xh[ot][v];
    return rsum_pair(s);
  };

  // Convert C-layout 64-dim activation (scaled by sc) into two B fragments (K=0..31, 32..63)
  auto buildB64 = [&](const float (&xh)[4][8], float sc, BFrag& B0, BFrag& B1) {
    unsigned own[4][4], prt[4][4];
    #pragma unroll
    for (int ot = 0; ot < 4; ++ot)
      #pragma unroll
      for (int k = 0; k < 4; ++k)
        own[ot][k] = pack2h(xh[ot][2 * k] * sc, xh[ot][2 * k + 1] * sc);
    #pragma unroll
    for (int ot = 0; ot < 4; ++ot)
      #pragma unroll
      for (int k = 0; k < 4; ++k)
        prt[ot][k] = __shfl_xor(own[ot][k], 16, 32);
    #pragma unroll
    for (int k = 0; k < 4; ++k) {
      B0.u[k]     = laneHi ? prt[1][k] : own[0][k];
      B0.u[4 + k] = laneHi ? own[1][k] : prt[0][k];
      B1.u[k]     = laneHi ? prt[3][k] : own[2][k];
      B1.u[4 + k] = laneHi ? own[3][k] : prt[2][k];
    }
  };

  // mobius_add(x, bias) for 64-dim x in C-layout; X2 = |x|^2 (already reduced)
  auto mobiusBias64 = [&](float (&xh)[4][8], const float* bp, float X2) {
    float bv[4][8]; float y2p = 0.f, xyp = 0.f;
    #pragma unroll
    for (int ot = 0; ot < 4; ++ot) {
      const v4f* q = (const v4f*)&bp[16 * ot + 8 * laneHi];
      v4f q0 = q[0], q1 = q[1];
      #pragma unroll
      for (int v = 0; v < 4; ++v) { bv[ot][v] = q0[v]; bv[ot][4 + v] = q1[v]; }
      #pragma unroll
      for (int v = 0; v < 8; ++v) { y2p += bv[ot][v] * bv[ot][v]; xyp += xh[ot][v] * bv[ot][v]; }
    }
    const float Y2 = rsum_pair(y2p), xy = rsum_pair(xyp);
    const float cx = 1.f + 2.f * xy + Y2, cy = 1.f - X2;
    const float idn = 1.f / fmaxf(1.f + 2.f * xy + X2 * Y2, 1e-15f);
    #pragma unroll
    for (int ot = 0; ot < 4; ++ot)
      #pragma unroll
      for (int v = 0; v < 8; ++v) xh[ot][v] = (cx * xh[ot][v] + cy * bv[ot][v]) * idn;
  };

  // ---------- main grid-stride tile loop (16 samples per wave-iteration) ----------
  for (int tile = blockIdx.x * WAVES_PER_BLOCK + wv; tile < nTiles;
       tile += (int)gridDim.x * WAVES_PER_BLOCK) {
    const int row  = tile * 16 + laneLo;
    const long rcl = (long)(row < nRows ? row : nRows - 1);

    // ---- one concat branch: h = expmap0(logmap0(x) @ Wc); returns tanh(|u|) = |h| ----
    float h[4][8], hb2[4][8];
    auto branchEval = [&](const float* X, int fbase, float (&hres)[4][8]) -> float {
      BFrag B;
      {
        const v4f* p = (const v4f*)(X + rcl * 32 + laneHi * 16);
        v4f q[4] = { p[0], p[1], p[2], p[3] };
        float vv[16]; float ss = 0.f;
        #pragma unroll
        for (int j = 0; j < 16; ++j) { vv[j] = q[j >> 2][j & 3]; ss += vv[j] * vv[j]; }
        ss = rsum_pair(ss);
        const float n  = sqrtf(ss);
        const float sc = atanh_fast(fminf(n, 0.99999f)) / fmaxf(n, 1e-15f);
        #pragma unroll
        for (int j = 0; j < 16; ++j) B.h[j] = (_Float16)(vv[j] * sc);
      }
      float ss = 0.f;
      #pragma unroll
      for (int ot = 0; ot < 4; ++ot) {
        v8f c = {};
        c = __builtin_amdgcn_wmma_f32_16x16x32_f16(false, loadA(fbase + ot), false, B.h,
                                                   (short)0, c, false, false);
        #pragma unroll
        for (int v = 0; v < 8; ++v) { hres[ot][v] = c[v]; ss += c[v] * c[v]; }
      }
      ss = rsum_pair(ss);
      const float n  = fmaxf(sqrtf(ss), 1e-15f);
      const float t  = tanh_pos(n);
      const float sc = t / n;
      #pragma unroll
      for (int ot = 0; ot < 4; ++ot)
        #pragma unroll
        for (int v = 0; v < 8; ++v) hres[ot][v] *= sc;
      return t;
    };

    const float t1 = branchEval(x1, 0, h);
    const float t2 = branchEval(x2, 4, hb2);

    // ---- h = mobius_add(h1, h2)  (|h1|^2 = t1^2, |h2|^2 = t2^2 analytically) ----
    {
      float xyp = 0.f;
      #pragma unroll
      for (int ot = 0; ot < 4; ++ot)
        #pragma unroll
        for (int v = 0; v < 8; ++v) xyp += h[ot][v] * hb2[ot][v];
      const float xy = rsum_pair(xyp);
      const float X2 = t1 * t1, Y2 = t2 * t2;
      const float cx = 1.f + 2.f * xy + Y2, cy = 1.f - X2;
      const float idn = 1.f / fmaxf(1.f + 2.f * xy + X2 * Y2, 1e-15f);
      #pragma unroll
      for (int ot = 0; ot < 4; ++ot)
        #pragma unroll
        for (int v = 0; v < 8; ++v) h[ot][v] = (cx * h[ot][v] + cy * hb2[ot][v]) * idn;
    }

    // ---- h = mobius_add(h, bc) ----
    mobiusBias64(h, &biasLDS[0], sumsq64(h));

    // ---- layer 0: h = mobius_add(expmap0(logmap0(h) @ W0), b0) ----
    {
      const float ss = sumsq64(h);
      const float n  = sqrtf(ss);
      const float sc = atanh_fast(fminf(n, 0.99999f)) / fmaxf(n, 1e-15f);
      BFrag B0, B1;
      buildB64(h, sc, B0, B1);
      float ss2 = 0.f;
      #pragma unroll
      for (int ot = 0; ot < 4; ++ot) {
        v8f c = {};
        c = __builtin_amdgcn_wmma_f32_16x16x32_f16(false, loadA(8 + ot),  false, B0.h, (short)0, c, false, false);
        c = __builtin_amdgcn_wmma_f32_16x16x32_f16(false, loadA(12 + ot), false, B1.h, (short)0, c, false, false);
        #pragma unroll
        for (int v = 0; v < 8; ++v) { h[ot][v] = c[v]; ss2 += c[v] * c[v]; }
      }
      ss2 = rsum_pair(ss2);
      const float n0  = fmaxf(sqrtf(ss2), 1e-15f);
      const float t0  = tanh_pos(n0);
      const float sc0 = t0 / n0;
      #pragma unroll
      for (int ot = 0; ot < 4; ++ot)
        #pragma unroll
        for (int v = 0; v < 8; ++v) h[ot][v] *= sc0;
      mobiusBias64(h, &biasLDS[64], t0 * t0);
    }

    // ---- layer 1: g = mobius_add(expmap0(logmap0(h) @ W1), b1)  (64 -> 32) ----
    float g[2][8];
    {
      const float ss = sumsq64(h);
      const float n  = sqrtf(ss);
      const float sc = atanh_fast(fminf(n, 0.99999f)) / fmaxf(n, 1e-15f);
      BFrag B0, B1;
      buildB64(h, sc, B0, B1);
      float ss2 = 0.f;
      #pragma unroll
      for (int ot = 0; ot < 2; ++ot) {
        v8f c = {};
        c = __builtin_amdgcn_wmma_f32_16x16x32_f16(false, loadA(16 + ot), false, B0.h, (short)0, c, false, false);
        c = __builtin_amdgcn_wmma_f32_16x16x32_f16(false, loadA(18 + ot), false, B1.h, (short)0, c, false, false);
        #pragma unroll
        for (int v = 0; v < 8; ++v) { g[ot][v] = c[v]; ss2 += c[v] * c[v]; }
      }
      ss2 = rsum_pair(ss2);
      const float n1  = fmaxf(sqrtf(ss2), 1e-15f);
      const float tg  = tanh_pos(n1);
      const float scg = tg / n1;
      #pragma unroll
      for (int ot = 0; ot < 2; ++ot)
        #pragma unroll
        for (int v = 0; v < 8; ++v) g[ot][v] *= scg;

      // mobius_add(g, b1): 32-dim
      float bv[2][8]; float y2p = 0.f, xyp = 0.f;
      #pragma unroll
      for (int ot = 0; ot < 2; ++ot) {
        const v4f* q = (const v4f*)&biasLDS[128 + 16 * ot + 8 * laneHi];
        v4f q0 = q[0], q1 = q[1];
        #pragma unroll
        for (int v = 0; v < 4; ++v) { bv[ot][v] = q0[v]; bv[ot][4 + v] = q1[v]; }
        #pragma unroll
        for (int v = 0; v < 8; ++v) { y2p += bv[ot][v] * bv[ot][v]; xyp += g[ot][v] * bv[ot][v]; }
      }
      const float Y2 = rsum_pair(y2p), xy = rsum_pair(xyp);
      const float X2 = tg * tg;
      const float cx = 1.f + 2.f * xy + Y2, cy = 1.f - X2;
      const float idn = 1.f / fmaxf(1.f + 2.f * xy + X2 * Y2, 1e-15f);
      #pragma unroll
      for (int ot = 0; ot < 2; ++ot)
        #pragma unroll
        for (int v = 0; v < 8; ++v) g[ot][v] = (cx * g[ot][v] + cy * bv[ot][v]) * idn;
    }

    // ---- store: out[row][m], m = 16*ot + 8*laneHi + v ; 2x32B per lane ----
    if (row < nRows) {
      float* po = out + (long)row * 32 + 8 * laneHi;
      v4f s0 = { g[0][0], g[0][1], g[0][2], g[0][3] };
      v4f s1 = { g[0][4], g[0][5], g[0][6], g[0][7] };
      v4f s2 = { g[1][0], g[1][1], g[1][2], g[1][3] };
      v4f s3 = { g[1][4], g[1][5], g[1][6], g[1][7] };
      *(v4f*)(po)      = s0;
      *(v4f*)(po + 4)  = s1;
      *(v4f*)(po + 16) = s2;
      *(v4f*)(po + 20) = s3;
    }
  }
}

extern "C" void kernel_launch(void* const* d_in, const int* in_sizes, int n_in,
                              void* d_out, int out_size, void* d_ws, size_t ws_size,
                              hipStream_t stream) {
  (void)n_in; (void)out_size; (void)d_ws; (void)ws_size;
  const float* x1  = (const float*)d_in[0];
  const float* x2  = (const float*)d_in[1];
  const float* Wc1 = (const float*)d_in[2];
  const float* Wc2 = (const float*)d_in[3];
  const float* bc  = (const float*)d_in[4];
  const float* W0  = (const float*)d_in[5];
  const float* b0  = (const float*)d_in[6];
  const float* W1  = (const float*)d_in[7];
  const float* b1  = (const float*)d_in[8];
  float* out = (float*)d_out;

  const int nRows  = in_sizes[0] / 32;
  const int nTiles = (nRows + 15) / 16;
  int blocks = (nTiles + WAVES_PER_BLOCK * TILES_PER_WAVE - 1) / (WAVES_PER_BLOCK * TILES_PER_WAVE);
  if (blocks < 1) blocks = 1;

  hipLaunchKernelGGL(poincare_mlp_wmma, dim3(blocks), dim3(THREADS), 0, stream,
                     x1, x2, Wc1, Wc2, bc, W0, b0, W1, b1, out, nRows);
}
